// UpResblock_65798898974753
// MI455X (gfx1250) — matrix operand order repaired and checked by reading, not compiled, not run on hardware
//
#include <hip/hip_runtime.h>

// ---------------------------------------------------------------------------
// UpResBlock (4D U-Net up block) for MI455X / gfx1250, bf16 WMMA implicit GEMM.
//
// Pipeline (all on `stream`):
//   K0 prep_weights : w1/w2/w3 fp32 -> bf16 repack [tap][cout][cin]
//   Kz zero_halo    : zero w-halo rows of xc and Y1 (EXEC-clean padding)
//   K1 build_xc     : xc = concat(x2, tconv2x2s2(x1,up_w)+up_b) channels-last bf16
//   K2 conv1_wmma   : Y1 = relu(conv3^4(xc)*w1 + b1)        (bf16, v_wmma)
//   K3 conv2sc_wmma : out = relu(conv3^4(Y1)*w2 + b2 + xc*w3 + b3)  (fp32 NCHW)
//
// xc / Y1 use an 18-wide padded w axis (w = -1..16 -> 0..17, zero halo) so
// every A-operand load is in-bounds: no per-lane predication, EXEC stays
// all-1s at every v_wmma (ISA 7.12 requirement).
//
// 4 independent f32 accumulators per wave break the WMMA->WMMA RAW chain
// (bf16 needs 1 NOP + 4 co-exec slots); the K-reduction split is summed in
// the epilogue.
//
// Workspace (~28.8 MB):
//   [0)      xc  : 2*4096*18*64 bf16 = 18.9 MB   (line-major, 18-w halo)
//   [+xc)    Y1  : 2*4096*18*32 bf16 =  9.4 MB
//   [+y1)    wb1 : 81*32*64 bf16, wb2: 81*32*32 bf16, wb3: 32*64 bf16
// ---------------------------------------------------------------------------

typedef __attribute__((ext_vector_type(16))) __bf16 bf16x16;
typedef __attribute__((ext_vector_type(8)))  __bf16 bf16x8;
typedef __attribute__((ext_vector_type(8)))  float  f32x8;

#define DEV static __device__ __forceinline__

#define WP 18                 // padded w extent
#define NLINE (2 * 4096)      // (b,t,d,h) lines

DEV unsigned short f2bf_u(float f) {            // RNE fp32->bf16
  unsigned u = __float_as_uint(f);
  u += 0x7fffu + ((u >> 16) & 1u);
  return (unsigned short)(u >> 16);
}

DEV bf16x16 pack16(bf16x8 lo, bf16x8 hi) {
  union { bf16x16 v; bf16x8 h[2]; } u;
  u.h[0] = lo; u.h[1] = hi;
  return u.v;
}
DEV bf16x16 load16(const __bf16* p0, const __bf16* p1) {
  return pack16(*(const bf16x8*)p0, *(const bf16x8*)p1);
}
DEV f32x8 zero_c() {
  f32x8 a;
#pragma unroll
  for (int i = 0; i < 8; ++i) a[i] = 0.0f;
  return a;
}
DEV f32x8 wmma_bf16(bf16x16 a, bf16x16 b, f32x8 c) {
  // D = A(16x32 bf16) * B(32x16 bf16) + C(16x16 f32)
  return __builtin_amdgcn_wmma_f32_16x16x32_bf16(
      /*neg_a=*/false, a, /*neg_b=*/false, b,
      /*c_mod=*/(short)0, c, /*reuse_a=*/false, /*reuse_b=*/false);
}
DEV float relu(float v) { return v > 0.f ? v : 0.f; }

// ---------------- K0: weight repack fp32 -> bf16 [tap][cout][cin] -----------
__global__ __launch_bounds__(256) void upres_prep_weights(
    const float* __restrict__ w1, const float* __restrict__ w2,
    const float* __restrict__ w3, unsigned short* __restrict__ wb1,
    unsigned short* __restrict__ wb2, unsigned short* __restrict__ wb3) {
  int i = blockIdx.x * 256 + threadIdx.x;
  if (i < 81 * 32 * 64) {                       // wb1[tap][co][ci], w1[co][ci][tap]
    int tap = i >> 11, co = (i >> 6) & 31, ci = i & 63;
    wb1[i] = f2bf_u(w1[(co * 64 + ci) * 81 + tap]);
  }
  if (i < 81 * 32 * 32) {                       // wb2[tap][co][ci], w2[co][ci][tap]
    int tap = i >> 10, co = (i >> 5) & 31, ci = i & 31;
    wb2[i] = f2bf_u(w2[(co * 32 + ci) * 81 + tap]);
  }
  if (i < 32 * 64) wb3[i] = f2bf_u(w3[i]);      // already [co][ci]
}

// ---------------- Kz: zero the w-halo rows of xc and Y1 ---------------------
__global__ __launch_bounds__(256) void upres_zero_halo(
    unsigned short* __restrict__ xc, unsigned short* __restrict__ y1) {
  int i = blockIdx.x * 256 + threadIdx.x;       // NLINE*64 threads
  int line = i >> 6, c = i & 63;
  size_t xb = (size_t)line * (WP * 64);
  xc[xb + c] = 0;                               // w = -1 row
  xc[xb + 17 * 64 + c] = 0;                     // w = 16 row
  if (c < 32) {
    size_t yb = (size_t)line * (WP * 32);
    y1[yb + c] = 0;
    y1[yb + 17 * 32 + c] = 0;
  }
}

// ---------------- K1: xc = concat(x2, tconv(x1)) channels-last bf16 ---------
__global__ __launch_bounds__(256) void upres_build_xc(
    const float* __restrict__ x1, const float* __restrict__ x2,
    const float* __restrict__ up_w, const float* __restrict__ up_b,
    unsigned short* __restrict__ xc) {
  int pos = blockIdx.x * 256 + threadIdx.x;     // 0 .. 131071  (b,t,d,h,w)
  int b = pos >> 16, sp = pos & 65535;
  int t = sp >> 12, d = (sp >> 8) & 15, h = (sp >> 4) & 15, w = sp & 15;
  int line = pos >> 4;                          // (b,t,d,h) line index
  unsigned short* dst = xc + (size_t)(line * WP + w + 1) * 64;

  // channels 0..31 <- x2 (NCHW fp32 -> channels-last bf16)
  const float* x2b = x2 + (size_t)b * 32 * 65536 + sp;
#pragma unroll
  for (int c = 0; c < 32; ++c) dst[c] = f2bf_u(x2b[(size_t)c << 16]);

  // channels 32..63 <- transposed conv k=2 s=2 (non-overlapping: 1 tap/output)
  int tap = ((t & 1) << 3) | ((d & 1) << 2) | ((h & 1) << 1) | (w & 1);
  int s1 = ((t >> 1) << 9) | ((d >> 1) << 6) | ((h >> 1) << 3) | (w >> 1);
  const float* x1b = x1 + (size_t)b * 64 * 4096 + s1;
  float acc[32];
#pragma unroll
  for (int co = 0; co < 32; ++co) acc[co] = up_b[co];
  for (int ci = 0; ci < 64; ++ci) {
    float xv = x1b[(size_t)ci * 4096];
    const float* wr = up_w + ((size_t)ci * 32) * 16 + tap;   // up_w[ci][co][tap]
#pragma unroll
    for (int co = 0; co < 32; ++co) acc[co] += xv * wr[co * 16];
  }
#pragma unroll
  for (int co = 0; co < 32; ++co) dst[32 + co] = f2bf_u(acc[co]);
}

// ---------------- K2: conv1 (3^4, 64->32) + bias + relu, bf16 WMMA ----------
__global__ __launch_bounds__(256) void upres_conv1_wmma(
    const __bf16* __restrict__ xc, const __bf16* __restrict__ wb1,
    const float* __restrict__ b1, unsigned short* __restrict__ y1) {
  const int lane = threadIdx.x & 31;
  // per-wave tile id, forced into SGPRs -> scalar tap-skip branches
  const int tile = __builtin_amdgcn_readfirstlane(blockIdx.x * 8 + (threadIdx.x >> 5));
  const int b = tile >> 12, t = (tile >> 8) & 15, d = (tile >> 4) & 15, h = tile & 15;
  const int col  = lane & 15;   // N column == cout lane; also D-matrix N
  const int half = lane >> 4;   // K-half select per ISA A/B layouts

  // 4 independent accumulators: [K-chunk][N-tile]  (summed at the end)
  f32x8 acc[2][2];
  acc[0][0] = zero_c(); acc[0][1] = zero_c();
  acc[1][0] = zero_c(); acc[1][1] = zero_c();

  for (int kt = 0; kt < 3; ++kt) {
    int tt = t + kt - 1; if ((unsigned)tt >= 16u) continue;     // scalar branch
    for (int kd = 0; kd < 3; ++kd) {
      int dd = d + kd - 1; if ((unsigned)dd >= 16u) continue;
      for (int kh = 0; kh < 3; ++kh) {
        int hh = h + kh - 1; if ((unsigned)hh >= 16u) continue;
        int linN = (((b * 16 + tt) * 16 + dd) * 16 + hh);
        // base pointers once per kh; kw offsets fold into load immediates
        const __bf16* arow0 = xc + (size_t)(linN * WP + col) * 64;
        const __bf16* wt0 = wb1 + (size_t)(((kt * 3 + kd) * 3 + kh) * 3) * (32 * 64);
        __builtin_prefetch(wt0 + 3 * 32 * 64, 0, 3);   // next kh weight block
#pragma unroll
        for (int kw = 0; kw < 3; ++kw) {
          const __bf16* arow = arow0 + kw * 64;        // w halo keeps this in-bounds
          const __bf16* wt = wt0 + kw * (32 * 64);
#pragma unroll
          for (int c = 0; c < 2; ++c) {                // K chunks of 32 channels
            bf16x16 a = load16(arow + c * 32 + half * 16,
                               arow + c * 32 + half * 16 + 8);
            const __bf16* bp0 = wt + col * 64 + c * 32 + half * 16;
            const __bf16* bp1 = wt + (16 + col) * 64 + c * 32 + half * 16;
            acc[c][0] = wmma_bf16(a, load16(bp0, bp0 + 8), acc[c][0]);
            acc[c][1] = wmma_bf16(a, load16(bp1, bp1 + 8), acc[c][1]);
          }
        }
      }
    }
  }
  // epilogue: sum chunk accumulators, +bias, relu, store bf16 into padded Y1
  // D layout: element v -> M = half*8+v (w coordinate), N = col (cout)
  float bias0 = b1[col], bias1 = b1[16 + col];
  unsigned short* outp = y1 + (size_t)(tile * WP + 1) * 32;
#pragma unroll
  for (int v = 0; v < 8; ++v) {
    int M = half * 8 + v;
    outp[(size_t)M * 32 + col]      = f2bf_u(relu(acc[0][0][v] + acc[1][0][v] + bias0));
    outp[(size_t)M * 32 + 16 + col] = f2bf_u(relu(acc[0][1][v] + acc[1][1][v] + bias1));
  }
}

// -------- K3: conv2 (3^4, 32->32) + skip 1x1 (64->32) + biases + relu -------
__global__ __launch_bounds__(256) void upres_conv2sc_wmma(
    const __bf16* __restrict__ y1, const __bf16* __restrict__ xc,
    const __bf16* __restrict__ wb2, const __bf16* __restrict__ wb3,
    const float* __restrict__ b2, const float* __restrict__ b3,
    float* __restrict__ out) {
  const int lane = threadIdx.x & 31;
  const int tile = __builtin_amdgcn_readfirstlane(blockIdx.x * 8 + (threadIdx.x >> 5));
  const int b = tile >> 12, t = (tile >> 8) & 15, d = (tile >> 4) & 15, h = tile & 15;
  const int col  = lane & 15;
  const int half = lane >> 4;

  // 4 independent accumulators: [parity][N-tile]
  f32x8 acc[2][2];
  acc[0][0] = zero_c(); acc[0][1] = zero_c();
  acc[1][0] = zero_c(); acc[1][1] = zero_c();

  // skip path: 1x1 conv on xc at own position (K = 64 -> chunk c uses parity c)
  {
    const __bf16* arow = xc + (size_t)(tile * WP + col + 1) * 64;
#pragma unroll
    for (int c = 0; c < 2; ++c) {
      bf16x16 a = load16(arow + c * 32 + half * 16, arow + c * 32 + half * 16 + 8);
      const __bf16* bp0 = wb3 + col * 64 + c * 32 + half * 16;
      const __bf16* bp1 = wb3 + (16 + col) * 64 + c * 32 + half * 16;
      acc[c][0] = wmma_bf16(a, load16(bp0, bp0 + 8), acc[c][0]);
      acc[c][1] = wmma_bf16(a, load16(bp1, bp1 + 8), acc[c][1]);
    }
  }
  // conv2: 81 taps, K = 32 per tap; alternate accumulator parity per tap
  int p = 0;
  for (int kt = 0; kt < 3; ++kt) {
    int tt = t + kt - 1; if ((unsigned)tt >= 16u) continue;
    for (int kd = 0; kd < 3; ++kd) {
      int dd = d + kd - 1; if ((unsigned)dd >= 16u) continue;
      for (int kh = 0; kh < 3; ++kh) {
        int hh = h + kh - 1; if ((unsigned)hh >= 16u) continue;
        int linN = (((b * 16 + tt) * 16 + dd) * 16 + hh);
        const __bf16* arow0 = y1 + (size_t)(linN * WP + col) * 32;
        const __bf16* wt0 = wb2 + (size_t)(((kt * 3 + kd) * 3 + kh) * 3) * (32 * 32);
        __builtin_prefetch(wt0 + 3 * 32 * 32, 0, 3);   // next kh weight block
#pragma unroll
        for (int kw = 0; kw < 3; ++kw) {
          const __bf16* arow = arow0 + kw * 32;
          bf16x16 a = load16(arow + half * 16, arow + half * 16 + 8);
          const __bf16* wt = wt0 + kw * (32 * 32);
          const __bf16* bp0 = wt + col * 32 + half * 16;
          const __bf16* bp1 = wt + (16 + col) * 32 + half * 16;
          acc[p][0] = wmma_bf16(a, load16(bp0, bp0 + 8), acc[p][0]);
          acc[p][1] = wmma_bf16(a, load16(bp1, bp1 + 8), acc[p][1]);
          p ^= 1;
        }
      }
    }
  }
  // epilogue: sum parities, +b2+b3, relu, vectorized fp32 NCHW store
  // out[(b*32+co)*65536 + spatial]; per lane: 8 consecutive w -> 2x b128 each
  float bias0 = b2[col] + b3[col];
  float bias1 = b2[16 + col] + b3[16 + col];
  int sp = (tile & 4095) << 4;                  // t*4096+d*256+h*16 within batch
  float* o0 = out + ((size_t)(b * 32 + col) << 16) + sp + half * 8;
  float* o1 = out + ((size_t)(b * 32 + 16 + col) << 16) + sp + half * 8;
  float4 s;
  s.x = relu(acc[0][0][0] + acc[1][0][0] + bias0);
  s.y = relu(acc[0][0][1] + acc[1][0][1] + bias0);
  s.z = relu(acc[0][0][2] + acc[1][0][2] + bias0);
  s.w = relu(acc[0][0][3] + acc[1][0][3] + bias0);
  *(float4*)o0 = s;
  s.x = relu(acc[0][0][4] + acc[1][0][4] + bias0);
  s.y = relu(acc[0][0][5] + acc[1][0][5] + bias0);
  s.z = relu(acc[0][0][6] + acc[1][0][6] + bias0);
  s.w = relu(acc[0][0][7] + acc[1][0][7] + bias0);
  *(float4*)(o0 + 4) = s;
  s.x = relu(acc[0][1][0] + acc[1][1][0] + bias1);
  s.y = relu(acc[0][1][1] + acc[1][1][1] + bias1);
  s.z = relu(acc[0][1][2] + acc[1][1][2] + bias1);
  s.w = relu(acc[0][1][3] + acc[1][1][3] + bias1);
  *(float4*)o1 = s;
  s.x = relu(acc[0][1][4] + acc[1][1][4] + bias1);
  s.y = relu(acc[0][1][5] + acc[1][1][5] + bias1);
  s.z = relu(acc[0][1][6] + acc[1][1][6] + bias1);
  s.w = relu(acc[0][1][7] + acc[1][1][7] + bias1);
  *(float4*)(o1 + 4) = s;
}

// ---------------------------------------------------------------------------
extern "C" void kernel_launch(void* const* d_in, const int* in_sizes, int n_in,
                              void* d_out, int out_size, void* d_ws, size_t ws_size,
                              hipStream_t stream) {
  const float* x1   = (const float*)d_in[0];
  const float* x2   = (const float*)d_in[1];
  const float* up_w = (const float*)d_in[2];
  const float* up_b = (const float*)d_in[3];
  const float* w1   = (const float*)d_in[4];
  const float* b1   = (const float*)d_in[5];
  const float* w2   = (const float*)d_in[6];
  const float* b2   = (const float*)d_in[7];
  const float* w3   = (const float*)d_in[8];
  const float* b3   = (const float*)d_in[9];
  float* out = (float*)d_out;

  char* ws = (char*)d_ws;
  const size_t XC_BYTES  = (size_t)NLINE * WP * 64 * 2;   // 18.9 MB
  const size_t Y1_BYTES  = (size_t)NLINE * WP * 32 * 2;   //  9.4 MB
  const size_t WB1_BYTES = (size_t)81 * 32 * 64 * 2;
  const size_t WB2_BYTES = (size_t)81 * 32 * 32 * 2;
  unsigned short* xc_w  = (unsigned short*)(ws);
  unsigned short* y1_w  = (unsigned short*)(ws + XC_BYTES);
  unsigned short* wb1_w = (unsigned short*)(ws + XC_BYTES + Y1_BYTES);
  unsigned short* wb2_w = (unsigned short*)(ws + XC_BYTES + Y1_BYTES + WB1_BYTES);
  unsigned short* wb3_w = (unsigned short*)(ws + XC_BYTES + Y1_BYTES + WB1_BYTES + WB2_BYTES);

  const __bf16* xc_r  = (const __bf16*)xc_w;
  const __bf16* y1_r  = (const __bf16*)y1_w;
  const __bf16* wb1_r = (const __bf16*)wb1_w;
  const __bf16* wb2_r = (const __bf16*)wb2_w;
  const __bf16* wb3_r = (const __bf16*)wb3_w;

  upres_prep_weights<<<(81 * 32 * 64 + 255) / 256, 256, 0, stream>>>(
      w1, w2, w3, wb1_w, wb2_w, wb3_w);
  upres_zero_halo<<<(NLINE * 64) / 256, 256, 0, stream>>>(xc_w, y1_w);
  upres_build_xc<<<131072 / 256, 256, 0, stream>>>(x1, x2, up_w, up_b, xc_w);
  upres_conv1_wmma<<<1024, 256, 0, stream>>>(xc_r, wb1_r, b1, y1_w);
  upres_conv2sc_wmma<<<1024, 256, 0, stream>>>(y1_r, xc_r, wb2_r, wb3_r, b2, b3, out);
}
